// TrustGNN_266287972651
// MI455X (gfx1250) — matrix operand order repaired and checked by reading, not compiled, not run on hardware
//
#include <hip/hip_runtime.h>
#include <hip/hip_fp16.h>
#include <stdint.h>

typedef __attribute__((ext_vector_type(16))) _Float16 v16h;
typedef __attribute__((ext_vector_type(8)))  _Float16 v8h;
typedef __attribute__((ext_vector_type(8)))  float    v8f;
typedef __attribute__((ext_vector_type(4)))  uint32_t u32x4;
typedef __attribute__((ext_vector_type(8)))  uint32_t u32x8;

// ---------------------------------------------------------------------------
// WMMA helpers: V_WMMA_F32_16X16X32_F16, wave32, layouts per CDNA5 ISA 7.12.2
// ---------------------------------------------------------------------------
__device__ inline v8f wmma16(v16h a, v16h b, v8f c) {
  return __builtin_amdgcn_wmma_f32_16x16x32_f16(
      /*neg_a=*/false, a, /*neg_b=*/false, b,
      /*c_mod=*/(short)0, c, /*reuse_a=*/false, /*reuse_b=*/false);
}

// A fragment (16x32 f16) from a row-major LDS tile; ld (halfs) keeps rows 16B
// aligned. lane<16: M=lane, K={0..7,16..23}; lane>=16: K+=8.
__device__ inline v16h load_a_frag(const _Float16* p, int lane, int ld) {
  const int m  = lane & 15;
  const int kg = (lane >> 4) * 8;
  const _Float16* rp = p + m * ld + kg;
  v8h lo = *(const v8h*)rp;          // K = kg + 0..7
  v8h hi = *(const v8h*)(rp + 16);   // K = kg + 16..23
  return __builtin_shufflevector(lo, hi, 0, 1, 2, 3, 4, 5, 6, 7,
                                 8, 9, 10, 11, 12, 13, 14, 15);
}

// B fragment from pre-swizzled weights (LDS): lane record contiguous (32B).
__device__ inline v16h load_b_swz(const _Float16* sW, int frag, int lane) {
  const _Float16* rp = sW + ((size_t)frag * 32 + lane) * 16;
  v8h lo = *(const v8h*)rp;
  v8h hi = *(const v8h*)(rp + 8);
  return __builtin_shufflevector(lo, hi, 0, 1, 2, 3, 4, 5, 6, 7,
                                 8, 9, 10, 11, 12, 13, 14, 15);
}

// B fragment streamed from pre-swizzled weights in global memory (L2-hot):
// two coalesced global_load_b128 per fragment.
__device__ inline v16h load_b_swz_g(const _Float16* __restrict__ gW, int frag, int lane) {
  const _Float16* rp = gW + ((size_t)frag * 32 + lane) * 16;
  v8h lo = *(const v8h*)rp;
  v8h hi = *(const v8h*)(rp + 8);
  return __builtin_shufflevector(lo, hi, 0, 1, 2, 3, 4, 5, 6, 7,
                                 8, 9, 10, 11, 12, 13, 14, 15);
}

__device__ inline uint32_t pack2(float v0, float v1) {
  union { _Float16 h[2]; uint32_t u; } pk;
  pk.h[0] = (_Float16)v0;
  pk.h[1] = (_Float16)v1;
  return pk.u;
}

// ---------------------------------------------------------------------------
// Tensor Data Mover: DMA a 1-D block of `nbytes` (multiple of 8) from global
// memory into LDS at byte offset `ldsByteOff`. D# per CDNA5 ISA ch.8:
//   group0: count=1 | lds_addr | global_addr[56:0] | type=2
//   group1: data_size=8B, tensor_dim0 = tile_dim0 = stride = nbytes/8, 1-D
// Issue once per workgroup (wave 0); complete with s_wait_tensorcnt.
// ---------------------------------------------------------------------------
__device__ inline void tdm_load_to_lds(const void* gsrc, uint32_t ldsByteOff,
                                       uint32_t nbytes) {
  const uint64_t ga = (uint64_t)(uintptr_t)gsrc;
  const uint32_t glo = (uint32_t)__builtin_amdgcn_readfirstlane((int)(uint32_t)ga);
  const uint32_t ghi = (uint32_t)__builtin_amdgcn_readfirstlane((int)(uint32_t)(ga >> 32));
  const uint32_t lds = (uint32_t)__builtin_amdgcn_readfirstlane((int)ldsByteOff);
  const uint32_t nw  = (uint32_t)__builtin_amdgcn_readfirstlane((int)(nbytes >> 3));
  u32x4 g0;
  g0[0] = 1u;                                   // count=1, user mode
  g0[1] = lds;                                  // lds_addr (bytes)
  g0[2] = glo;                                  // global_addr[31:0]
  g0[3] = (ghi & 0x01FFFFFFu) | (2u << 30);     // global_addr[56:32] | type=2
  u32x8 g1;
  g1[0] = 3u << 16;                             // data_size = 8 bytes
  g1[1] = (nw & 0xFFFFu) << 16;                 // tensor_dim0[15:0]
  g1[2] = (nw >> 16) | (1u << 16);              // tensor_dim0[31:16] | tensor_dim1=1
  g1[3] = (nw & 0xFFFFu) << 16;                 // tile_dim0
  g1[4] = 0u;                                   // tile_dim1/2 unused
  g1[5] = nw;                                   // tensor_dim0_stride[31:0]
  g1[6] = 0u;                                   // stride hi / dim1_stride lo
  g1[7] = 0u;
  asm volatile("tensor_load_to_lds %0, %1" :: "s"(g0), "s"(g1) : "memory");
}

// ---------------------------------------------------------------------------
// Utility kernels
// ---------------------------------------------------------------------------
// Convert f32 weight [srcRows x srcCols] into padded [KP x NP] f16 stored in
// B-fragment-major order: dst[((kk*NP/16 + nt)*32 + lane)*16 + h] holds
// element (K = kk*32 + (lane>>4)*8 + (h<8 ? h : h+8), N = nt*16 + (lane&15)).
__global__ void cvt_swz_f16_kernel(const float* __restrict__ src, _Float16* __restrict__ dst,
                                   int srcRows, int srcCols, int KP, int NP) {
  int idx = blockIdx.x * blockDim.x + threadIdx.x;
  if (idx >= KP * NP) return;
  int f = idx >> 9;
  int rem = idx & 511;
  int l = rem >> 4;
  int h = rem & 15;
  int NT = NP >> 4;
  int kk = f / NT, nt = f % NT;
  int k = kk * 32 + (l >> 4) * 8 + (h < 8 ? h : h + 8);
  int n = nt * 16 + (l & 15);
  float v = (k < srcRows && n < srcCols) ? src[k * srcCols + n] : 0.0f;
  dst[idx] = (_Float16)v;
}

__global__ void zero_f32_kernel(float* __restrict__ p, size_t n) {
  size_t i = (size_t)blockIdx.x * blockDim.x + threadIdx.x;
  if (i < n) p[i] = 0.0f;
}

// ---------------------------------------------------------------------------
// Edge message-passing kernel (one layer):
//   hidden = relu(edge_attr @ Wm1 + bm1)        [16 x 288]  (K padded 129->160)
//   edge_t = hidden @ Wm2 + bm2                 [16 x DOUT] (K padded 258->288)
//   atomicAdd(aggr[col[e]], edge_t + nodeFeat[row[e]])
// Wm1 DMA'd into LDS by the TDM; Wm2 fragments streamed from global (L2-hot).
// 8 waves / WG, one 16-edge tile per wave.
// ---------------------------------------------------------------------------
template <int DOUT, int NWAVES>
__global__ __launch_bounds__(NWAVES * 32)
void edge_mp_kernel(const float* __restrict__ trust, const float* __restrict__ qe,
                    const long long* __restrict__ row, const long long* __restrict__ col,
                    const float* __restrict__ nodeFeat,
                    const _Float16* __restrict__ Wm1s,   // swz [160 x 288]
                    const _Float16* __restrict__ Wm2s,   // swz [288 x DOUT] (global)
                    const float* __restrict__ bm1, int bm1n,
                    const float* __restrict__ bm2,       // [DOUT]
                    float* __restrict__ aggr, int nEdges) {
  constexpr int K1 = 160, H = 288, AS = 168, HS = 296;
  constexpr int NT1 = H / 16, NT2 = DOUT / 16, KK1 = K1 / 32, KK2 = H / 32;
  extern __shared__ char smem[];
  _Float16* sW1   = (_Float16*)smem;                    // 160*288 halfs (swz)
  float*    sB1   = (float*)(sW1 + K1 * H);             // 288 f32
  float*    sB2   = sB1 + H;                            // DOUT f32
  _Float16* sAttr = (_Float16*)(sB2 + DOUT);            // NWAVES*16*AS
  _Float16* sHid  = sAttr + NWAVES * 16 * AS;           // NWAVES*16*HS

  const int tid = threadIdx.x;
  const int wave = tid >> 5;
  const int lane = tid & 31;

  if (wave == 0)
    tdm_load_to_lds(Wm1s, (uint32_t)(uintptr_t)sW1, (uint32_t)(K1 * H * 2));
  for (int i = tid; i < H; i += blockDim.x) sB1[i] = (i < bm1n) ? bm1[i] : 0.0f;
  for (int i = tid; i < DOUT; i += blockDim.x) sB2[i] = bm2[i];
  if (wave == 0) __builtin_amdgcn_s_wait_tensorcnt(0);
  __syncthreads();

  const int e0 = (blockIdx.x * NWAVES + wave) * 16;
  _Float16* A  = sAttr + wave * 16 * AS;
  _Float16* Hd = sHid + wave * 16 * HS;
  uint32_t* A32 = (uint32_t*)A;

  // Stage edge_attr tile [16 x 160] f16 (col0=trust, 1..128=qe, rest 0).
  for (int m = 0; m < 16; ++m) {
    const int e = e0 + m;
    const bool ok = (e < nEdges);
    for (int c2 = lane; c2 < K1 / 2; c2 += 32) {
      const int c = c2 * 2;
      float v0 = 0.0f, v1 = 0.0f;
      if (ok) {
        if (c == 0) {
          v0 = trust[e];
          v1 = qe[(size_t)e * 128];
        } else if (c < 129) {
          v0 = qe[(size_t)e * 128 + (c - 1)];
          v1 = (c < 128) ? qe[(size_t)e * 128 + c] : 0.0f;
        }
      }
      A32[m * (AS / 2) + c2] = pack2(v0, v1);
    }
  }

  const int n  = lane & 15;
  const int mb = (lane >> 4) * 8;

  // GEMM1: hidden = relu(A @ W1 + b1), B from LDS
  v16h aF[KK1];
#pragma unroll
  for (int kk = 0; kk < KK1; ++kk) aF[kk] = load_a_frag(A + kk * 32, lane, AS);
#pragma unroll
  for (int nt = 0; nt < NT1; ++nt) {
    const float bias = sB1[nt * 16 + n];
    v8f c;
#pragma unroll
    for (int r = 0; r < 8; ++r) c[r] = bias;
#pragma unroll
    for (int kk = 0; kk < KK1; ++kk)
      c = wmma16(aF[kk], load_b_swz(sW1, kk * NT1 + nt, lane), c);
#pragma unroll
    for (int r = 0; r < 8; ++r) {
      float v = c[r] > 0.0f ? c[r] : 0.0f;
      Hd[(mb + r) * HS + nt * 16 + n] = (_Float16)v;
    }
  }

  // GEMM2 (B streamed from global) + gather(nodeFeat[row]) + scatter-add
  v16h hF[KK2];
#pragma unroll
  for (int kk = 0; kk < KK2; ++kk) hF[kk] = load_a_frag(Hd + kk * 32, lane, HS);
#pragma unroll
  for (int nt = 0; nt < NT2; ++nt) {
    const float bias = sB2[nt * 16 + n];
    v8f c;
#pragma unroll
    for (int r = 0; r < 8; ++r) c[r] = bias;
#pragma unroll
    for (int kk = 0; kk < KK2; ++kk)
      c = wmma16(hF[kk], load_b_swz_g(Wm2s, kk * NT2 + nt, lane), c);
#pragma unroll
    for (int r = 0; r < 8; ++r) {
      const int e = e0 + mb + r;
      if (e < nEdges) {
        const int rs = (int)row[e];
        const int cd = (int)col[e];
        const float v = c[r] + nodeFeat[(size_t)rs * DOUT + nt * 16 + n];
        unsafeAtomicAdd(&aggr[(size_t)cd * DOUT + nt * 16 + n], v);
      }
    }
  }
}

// ---------------------------------------------------------------------------
// Node update: out = relu(aggr @ Wl + bl); optional f32 and f16 outputs.
// Weight block DMA'd into LDS by the TDM.
// ---------------------------------------------------------------------------
template <int DIN, int DOUT, int NWAVES>
__global__ __launch_bounds__(NWAVES * 32)
void node_update_kernel(const float* __restrict__ aggr, const _Float16* __restrict__ Ws,
                        const float* __restrict__ bl, float* __restrict__ outF,
                        _Float16* __restrict__ outH, int nNodes) {
  constexpr int AS = DIN + 8, NT = DOUT / 16, KK = DIN / 32;
  extern __shared__ char smem[];
  _Float16* sW = (_Float16*)smem;          // DIN*DOUT halfs (swz)
  _Float16* sA = sW + DIN * DOUT;          // NWAVES*16*AS halfs
  const int tid = threadIdx.x;
  const int wave = tid >> 5;
  const int lane = tid & 31;

  if (wave == 0) {
    tdm_load_to_lds(Ws, (uint32_t)(uintptr_t)sW, (uint32_t)(DIN * DOUT * 2));
    __builtin_amdgcn_s_wait_tensorcnt(0);
  }
  __syncthreads();

  const int t0 = (blockIdx.x * NWAVES + wave) * 16;
  if (t0 >= nNodes) return;                // wave-uniform: EXEC stays all-1
  _Float16* A = sA + wave * 16 * AS;
  uint32_t* A32 = (uint32_t*)A;

  for (int m = 0; m < 16; ++m) {
    const int node = t0 + m;
    for (int c2 = lane; c2 < DIN / 2; c2 += 32) {
      float v0 = 0.0f, v1 = 0.0f;
      if (node < nNodes) {
        v0 = aggr[(size_t)node * DIN + 2 * c2];
        v1 = aggr[(size_t)node * DIN + 2 * c2 + 1];
      }
      A32[m * (AS / 2) + c2] = pack2(v0, v1);
    }
  }

  const int n  = lane & 15;
  const int mb = (lane >> 4) * 8;
  v16h aF[KK];
#pragma unroll
  for (int kk = 0; kk < KK; ++kk) aF[kk] = load_a_frag(A + kk * 32, lane, AS);
#pragma unroll
  for (int nt = 0; nt < NT; ++nt) {
    const float bias = bl[nt * 16 + n];
    v8f c;
#pragma unroll
    for (int r = 0; r < 8; ++r) c[r] = bias;
#pragma unroll
    for (int kk = 0; kk < KK; ++kk)
      c = wmma16(aF[kk], load_b_swz(sW, kk * NT + nt, lane), c);
#pragma unroll
    for (int r = 0; r < 8; ++r) {
      const int node = t0 + mb + r;
      if (node < nNodes) {
        float v = c[r] > 0.0f ? c[r] : 0.0f;
        if (outF) outF[(size_t)node * DOUT + nt * 16 + n] = v;
        if (outH) outH[(size_t)node * DOUT + nt * 16 + n] = (_Float16)v;
      }
    }
  }
}

// ---------------------------------------------------------------------------
// Prediction: sigmoid(relu([h2[row]|h2[col]|qe] @ Wp1 + bp1) @ Wp2 + bp2)
// Wp1 fragments streamed from global (L2-hot); LDS holds staging only.
// Final [256]-dot folded into the C fragments + lane shuffle reduction.
// ---------------------------------------------------------------------------
template <int NWAVES>
__global__ __launch_bounds__(NWAVES * 32)
void predict_kernel(const _Float16* __restrict__ h2h, const float* __restrict__ qe,
                    const long long* __restrict__ row, const long long* __restrict__ col,
                    const _Float16* __restrict__ Wp1s,  // swz [384 x 256] (global)
                    const float* __restrict__ bp1, const float* __restrict__ Wp2,
                    const float* __restrict__ bp2, float* __restrict__ out, int nEdges) {
  constexpr int K = 384, DH = 256, AS = 392, NT = DH / 16, KK = K / 32;
  extern __shared__ char smem[];
  _Float16* sA = (_Float16*)smem;          // NWAVES*16*AS halfs
  const int tid = threadIdx.x;
  const int wave = tid >> 5;
  const int lane = tid & 31;
  const int e0 = (blockIdx.x * NWAVES + wave) * 16;
  _Float16* A = sA + wave * 16 * AS;
  uint32_t* A32 = (uint32_t*)A;
  const uint32_t* h2u = (const uint32_t*)h2h;   // 2 halfs per u32, 64 per row

  for (int m = 0; m < 16; ++m) {
    const int e = e0 + m;
    const bool ok = (e < nEdges);
    const int rs = ok ? (int)row[e] : 0;
    const int cd = ok ? (int)col[e] : 0;
    for (int c2 = lane; c2 < K / 2; c2 += 32) {
      const int c = 2 * c2;
      uint32_t u = 0;
      if (ok) {
        if (c < 128)       u = h2u[(size_t)rs * 64 + c2];
        else if (c < 256)  u = h2u[(size_t)cd * 64 + (c2 - 64)];
        else               u = pack2(qe[(size_t)e * 128 + (c - 256)],
                                     qe[(size_t)e * 128 + (c - 255)]);
      }
      A32[m * (AS / 2) + c2] = u;
    }
  }

  const int n  = lane & 15;
  const int mb = (lane >> 4) * 8;
  v16h aF[KK];
#pragma unroll
  for (int kk = 0; kk < KK; ++kk) aF[kk] = load_a_frag(A + kk * 32, lane, AS);

  float part[8];
#pragma unroll
  for (int r = 0; r < 8; ++r) part[r] = 0.0f;

#pragma unroll
  for (int nt = 0; nt < NT; ++nt) {
    const float bias = bp1[nt * 16 + n];
    v8f c;
#pragma unroll
    for (int r = 0; r < 8; ++r) c[r] = bias;
#pragma unroll
    for (int kk = 0; kk < KK; ++kk)
      c = wmma16(aF[kk], load_b_swz_g(Wp1s, kk * NT + nt, lane), c);
    const float w2 = Wp2[nt * 16 + n];
#pragma unroll
    for (int r = 0; r < 8; ++r) {
      float h = c[r] > 0.0f ? c[r] : 0.0f;
      part[r] += h * w2;
    }
  }
  // reduce over the 16 lanes that share each row (xor of lane bits 0..3)
#pragma unroll
  for (int off = 1; off < 16; off <<= 1)
#pragma unroll
    for (int r = 0; r < 8; ++r) part[r] += __shfl_xor(part[r], off, 32);

  const float b2 = bp2[0];
  if ((lane & 15) < 8) {
    const int r = lane & 15;
    const int e = e0 + mb + r;
    if (e < nEdges) {
      const float v = part[r] + b2;
      out[e] = 1.0f / (1.0f + __expf(-v));
    }
  }
}

// ---------------------------------------------------------------------------
// Host driver
// ---------------------------------------------------------------------------
extern "C" void kernel_launch(void* const* d_in, const int* in_sizes, int n_in,
                              void* d_out, int out_size, void* d_ws, size_t ws_size,
                              hipStream_t stream) {
  const float*     x      = (const float*)d_in[0];
  const long long* ei     = (const long long*)d_in[1];
  const float*     trust  = (const float*)d_in[2];
  const float*     qe     = (const float*)d_in[3];
  const float *Wm1_1 = (const float*)d_in[4],  *bm1_1 = (const float*)d_in[5];
  const float *Wm2_1 = (const float*)d_in[6],  *bm2_1 = (const float*)d_in[7];
  const float *Wl_1  = (const float*)d_in[8],  *bl_1  = (const float*)d_in[9];
  const float *Wm1_2 = (const float*)d_in[10], *bm1_2 = (const float*)d_in[11];
  const float *Wm2_2 = (const float*)d_in[12], *bm2_2 = (const float*)d_in[13];
  const float *Wl_2  = (const float*)d_in[14], *bl_2  = (const float*)d_in[15];
  const float *Wp1   = (const float*)d_in[16], *bp1   = (const float*)d_in[17];
  const float *Wp2   = (const float*)d_in[18], *bp2   = (const float*)d_in[19];

  const int N = in_sizes[0] / 128;   // 50000
  const int E = in_sizes[2];         // 800000
  const long long* rowI = ei;
  const long long* colI = ei + E;

  // ---- workspace carve-out (256B aligned) ----
  char* ws = (char*)d_ws;
  size_t off = 0;
  auto take = [&](size_t bytes) -> char* {
    char* p = ws + off;
    off += (bytes + 255) & ~(size_t)255;
    return p;
  };
  _Float16* pW11 = (_Float16*)take((size_t)160 * 288 * 2);  // Wm1_1 swz
  _Float16* pW21 = (_Float16*)take((size_t)288 * 128 * 2);  // Wm2_1 swz
  _Float16* pW12 = (_Float16*)take((size_t)160 * 288 * 2);  // Wm1_2 swz
  _Float16* pW22 = (_Float16*)take((size_t)288 * 256 * 2);  // Wm2_2 swz
  _Float16* pWl1 = (_Float16*)take((size_t)128 * 256 * 2);
  _Float16* pWl2 = (_Float16*)take((size_t)256 * 128 * 2);
  _Float16* pWp1 = (_Float16*)take((size_t)384 * 256 * 2);
  float*    aggr1 = (float*)take((size_t)N * 128 * 4);
  float*    h1    = (float*)take((size_t)N * 256 * 4);
  float*    aggr2 = (float*)take((size_t)N * 256 * 4);
  _Float16* h2h   = (_Float16*)take((size_t)N * 128 * 2);

  auto cvt = [&](const float* s, _Float16* d, int sr, int sc, int kp, int np) {
    int tot = kp * np;
    cvt_swz_f16_kernel<<<(tot + 255) / 256, 256, 0, stream>>>(s, d, sr, sc, kp, np);
  };
  cvt(Wm1_1, pW11, 129, 258, 160, 288);
  cvt(Wm2_1, pW21, 258, 128, 288, 128);
  cvt(Wm1_2, pW12, 129, 258, 160, 288);
  cvt(Wm2_2, pW22, 258, 256, 288, 256);
  cvt(Wl_1,  pWl1, 128, 256, 128, 256);
  cvt(Wl_2,  pWl2, 256, 128, 256, 128);
  cvt(Wp1,   pWp1, 384, 256, 384, 256);

  {
    size_t n1 = (size_t)N * 128, n2 = (size_t)N * 256;
    zero_f32_kernel<<<(int)((n1 + 255) / 256), 256, 0, stream>>>(aggr1, n1);
    zero_f32_kernel<<<(int)((n2 + 255) / 256), 256, 0, stream>>>(aggr2, n2);
  }

  // ---- LDS footprints ----
  auto edgeLds = [](int dout, int nw) -> size_t {
    return (size_t)160 * 288 * 2 + (288 + dout) * 4 +
           (size_t)nw * 16 * 168 * 2 + (size_t)nw * 16 * 296 * 2;
  };
  const size_t lds_e1 = edgeLds(128, 8);                                  // ~213 KB
  const size_t lds_e2 = edgeLds(256, 8);                                  // ~213 KB
  const size_t lds_n1 = (size_t)128 * 256 * 2 + (size_t)8 * 16 * 136 * 2; // ~100 KB
  const size_t lds_n2 = (size_t)256 * 128 * 2 + (size_t)8 * 16 * 264 * 2; // ~133 KB
  const size_t lds_p  = (size_t)8 * 16 * 392 * 2;                         // ~100 KB

  const int edgeBlocks = ((E + 15) / 16 + 7) / 8;   // 6250
  const int nodeBlocks = ((N + 15) / 16 + 7) / 8;   // 391
  const int predBlocks = ((E + 15) / 16 + 7) / 8;   // 6250

  // Layer 1
  edge_mp_kernel<128, 8><<<edgeBlocks, 256, lds_e1, stream>>>(
      trust, qe, rowI, colI, x, pW11, pW21, bm1_1, 258, bm2_1, aggr1, E);
  node_update_kernel<128, 256, 8><<<nodeBlocks, 256, lds_n1, stream>>>(
      aggr1, pWl1, bl_1, h1, (_Float16*)nullptr, N);

  // Layer 2
  edge_mp_kernel<256, 8><<<edgeBlocks, 256, lds_e2, stream>>>(
      trust, qe, rowI, colI, h1, pW12, pW22, bm1_2, 258, bm2_2, aggr2, E);
  node_update_kernel<256, 128, 8><<<nodeBlocks, 256, lds_n2, stream>>>(
      aggr2, pWl2, bl_2, (float*)nullptr, h2h, N);

  // Per-edge prediction
  predict_kernel<8><<<predBlocks, 256, lds_p, stream>>>(
      h2h, qe, rowI, colI, pWp1, bp1, Wp2, bp2, (float*)d_out, E);
}
